// Net_50620484551136
// MI455X (gfx1250) — matrix operand order repaired and checked by reading, not compiled, not run on hardware
//
#include <hip/hip_runtime.h>
#include <hip/hip_bf16.h>

#define T_NODES 50000
#define N_BASE  5000
#define E_EDGES 800000
#define FIN     4652
#define NCLS    5

typedef __attribute__((ext_vector_type(16))) __bf16 v16bf;
typedef __attribute__((ext_vector_type(8)))  float  v8f;
typedef __attribute__((ext_vector_type(4)))  unsigned int uint4v;

union FragLd { v16bf v; uint4v q[2]; __bf16 e[16]; };
union Pack8  { uint4v q; __bf16 e[8]; };

// ---------------------------------------------------------------------------
// Pre-swizzle a weight matrix W[K,256] (f32, row-major) into bf16 WMMA
// B-fragment order:  Wsw[((kt*16 + nt)*32 + lane)*16 + e]
//   lane = n + 16*khalf  (n = column within 16-tile, khalf = K half of 32-tile)
//   e    = K offset within half:  K = kt*32 + khalf*16 + e
// A lane's whole fragment is then 32 contiguous bytes.
// ---------------------------------------------------------------------------
__global__ void swizzle_w(const float* __restrict__ W, __bf16* __restrict__ Wsw,
                          int K, int nslots) {
    int slot = blockIdx.x * 256 + threadIdx.x;
    if (slot >= nslots) return;
    const int kt   = slot >> 9;
    const int nt   = (slot >> 5) & 15;
    const int lane = slot & 31;
    const int n    = lane & 15;
    const int kh   = lane >> 4;
    const int kb   = kt * 32 + kh * 16;
    Pack8 p0, p1;
#pragma unroll
    for (int e = 0; e < 8; ++e) {
        int k0 = kb + e, k1 = kb + 8 + e;
        p0.e[e] = (__bf16)((k0 < K) ? W[(long)k0 * 256 + nt * 16 + n] : 0.0f);
        p1.e[e] = (__bf16)((k1 < K) ? W[(long)k1 * 256 + nt * 16 + n] : 0.0f);
    }
    uint4v* dst = (uint4v*)&Wsw[(long)slot * 16];
    dst[0] = p0.q;
    dst[1] = p1.q;
}

// ---------------------------------------------------------------------------
// WMMA GEMM: C[M,256] = act(A[M,K] @ W[K,256] + bias), W pre-swizzled bf16.
// Split-A: k < ksplit -> A1 (ld=ksplit), k >= ksplit -> A2 (ld=K-ksplit);
// A2==nullptr: all A1, ld=K.  Block = 256 thr (8 waves), tile 128x256.
// Per K-step: load ALL 16 B fragments (L2-fast, FIFO head), then issue the
// HBM staging loads for the next A tile (FIFO tail), then 16 back-to-back
// WMMAs with staged loadcnt waits.  A tiles double-buffered in LDS.
// ---------------------------------------------------------------------------
__launch_bounds__(256)
__global__ void gemm_wmma_bf16(const float* __restrict__ A1,
                               const float* __restrict__ A2, int ksplit,
                               const __bf16* __restrict__ Wsw,
                               const float* __restrict__ bias,
                               float* __restrict__ C,
                               int M, int K, int relu)
{
    __shared__ __bf16 sA[2][128 * 32];   // 2 x 8 KB, swizzled A tiles

    const int tid  = threadIdx.x;
    const int lane = tid & 31;
    const int wave = tid >> 5;
    const int row0 = blockIdx.x * 128;
    const int m16  = lane & 15;
    const int hi   = lane >> 4;
    const int KT   = (K + 31) >> 5;

    v8f acc[16];
#pragma unroll
    for (int i = 0; i < 16; ++i) acc[i] = v8f{};

    // stage A k-tile `kt` into buffer `buf`, swizzled to fragment order.
    // thread t: row r = t>>1, K-half h = t&1 -> 16 consecutive K values:
    //   i<8  -> sA[r*32 + h*16 + i]
    //   i>=8 -> sA[r*32 + h*16 + i]  (contiguous: LDS pos = K offset)
    auto stageA = [&](int kt, int buf) {
        const int r    = tid >> 1;
        const int h    = tid & 1;
        const int kb   = kt * 32 + h * 16;
        const int arow = row0 + r;
        const float* Ap; int lda, kof;
        if (A2 && kb >= ksplit) { Ap = A2; lda = K - ksplit; kof = kb - ksplit; }
        else                    { Ap = A1; lda = A2 ? ksplit : K; kof = kb; }
        const float* src = Ap + (long)arow * lda + kof;
        Pack8 p0, p1;
#pragma unroll
        for (int i = 0; i < 8; ++i) {
            p0.e[i] = (__bf16)((arow < M && (kb + i)     < K) ? src[i]     : 0.0f);
            p1.e[i] = (__bf16)((arow < M && (kb + 8 + i) < K) ? src[8 + i] : 0.0f);
        }
        *(uint4v*)&sA[buf][r * 32 + h * 16]     = p0.q;
        *(uint4v*)&sA[buf][r * 32 + h * 16 + 8] = p1.q;
    };

    if (KT > 0) stageA(0, 0);
    __syncthreads();

    for (int kt = 0; kt < KT; ++kt) {
        const int cur = kt & 1;

        // A fragment: 32 contiguous bytes -> 2x ds_load_b128 (DScnt, independent)
        FragLd afr;
        {
            const uint4v* ap = (const uint4v*)&sA[cur][(wave * 16 + m16) * 32 + hi * 16];
            afr.q[0] = ap[0];
            afr.q[1] = ap[1];
        }

        // All 16 B fragments into distinct registers: 32x global_load_b128
        // at the HEAD of the LOADcnt FIFO (fast, L2/WGP$-resident weights).
        const __bf16* wk = Wsw + (long)kt * 8192;
        FragLd bfr[16];
        {
            const uint4v* bp = (const uint4v*)(wk + lane * 16);
#pragma unroll
            for (int nt = 0; nt < 16; ++nt) {
                bfr[nt].q[0] = bp[nt * 64];       // 512 bf16 = 64 uint4v per tile
                bfr[nt].q[1] = bp[nt * 64 + 1];
            }
        }

        // Next A tile HBM loads go at the TAIL of the FIFO: nothing below
        // waits on them until the end-of-iteration barrier.
        if (kt + 1 < KT) stageA(kt + 1, cur ^ 1);

        // 16 back-to-back WMMAs (distinct accumulators, no D->A/B hazard).
#pragma unroll
        for (int nt = 0; nt < 16; ++nt)
            acc[nt] = __builtin_amdgcn_wmma_f32_16x16x32_bf16(
                false, afr.v, false, bfr[nt].v, (short)0, acc[nt], false, false);

        __syncthreads();
    }

    // epilogue: C layout VGPR r -> M = r (lanes 0-15) / r+8 (lanes 16-31)
    const int rbase = row0 + wave * 16 + hi * 8;
#pragma unroll
    for (int nt = 0; nt < 16; ++nt) {
        const int col = nt * 16 + m16;
        const float bv = bias ? bias[col] : 0.0f;
#pragma unroll
        for (int r = 0; r < 8; ++r) {
            const int row = rbase + r;
            if (row < M) {
                float v = acc[nt][r] + bv;
                if (relu) v = fmaxf(v, 0.0f);
                C[(long)row * 256 + col] = v;
            }
        }
    }
}

// ---------------------------------------------------------------------------
// Elementwise / graph kernels
// ---------------------------------------------------------------------------
__global__ void fill_f32(float* p, float v, int n) {
    int i = blockIdx.x * 256 + threadIdx.x;
    if (i < n) p[i] = v;
}

__global__ void deg_count(const int* __restrict__ dst, float* __restrict__ deg, int E) {
    int i = blockIdx.x * 256 + threadIdx.x;
    if (i < E) atomicAdd(&deg[dst[i]], 1.0f);
}

__global__ void rsqrt_ip(float* p, int n) {
    int i = blockIdx.x * 256 + threadIdx.x;
    if (i < n) { float d = p[i]; p[i] = (d > 0.0f) ? rsqrtf(d) : 0.0f; }
}

// agg[t,f] = hp[t,f] * dinv[t]^2   (self-loop term; initializes agg)
__global__ void gcn_self_init(const float* __restrict__ hp, const float* __restrict__ dinv,
                              float* __restrict__ agg) {
    int t = blockIdx.x, f = threadIdx.x;
    float d = dinv[t];
    agg[(long)t * 256 + f] = hp[(long)t * 256 + f] * d * d;
}

// one block per edge: agg[dst] += hp[src] * dinv[src]*dinv[dst]
__global__ void gcn_edge_scatter(const float* __restrict__ hp,
                                 const int* __restrict__ src, const int* __restrict__ dst,
                                 const float* __restrict__ dinv, float* __restrict__ agg) {
    int e = blockIdx.x;
    int s = src[e], d = dst[e];
    float nrm = dinv[s] * dinv[d];
    int f = threadIdx.x;
    atomicAdd(&agg[(long)d * 256 + f], hp[(long)s * 256 + f] * nrm);
}

__global__ void bias_relu256(float* __restrict__ h, const float* __restrict__ b) {
    int t = blockIdx.x, f = threadIdx.x;
    long i = (long)t * 256 + f;
    h[i] = fmaxf(h[i] + b[f], 0.0f);
}

__global__ void seg_count(const int* __restrict__ idx, float* __restrict__ cnt, int T) {
    int i = blockIdx.x * 256 + threadIdx.x;
    if (i < T) atomicAdd(&cnt[idx[i]], 1.0f);
}

__global__ void seg_sum256(const float* __restrict__ h, const int* __restrict__ idx,
                           float* __restrict__ out) {
    int t = blockIdx.x, f = threadIdx.x;
    atomicAdd(&out[(long)idx[t] * 256 + f], h[(long)t * 256 + f]);
}

__global__ void seg_div256(float* __restrict__ r, const float* __restrict__ cnt) {
    int n = blockIdx.x, f = threadIdx.x;
    r[(long)n * 256 + f] /= fmaxf(cnt[n], 1.0f);
}

// logits = feats @ W_fb[256,5] + b_fb ; out = log_softmax(logits)
__global__ void final_head(const float* __restrict__ feats, const float* __restrict__ Wfb,
                           const float* __restrict__ bfb, float* __restrict__ out, int Nn) {
    int n = blockIdx.x * 128 + threadIdx.x;
    if (n >= Nn) return;
    float acc[NCLS];
#pragma unroll
    for (int c = 0; c < NCLS; ++c) acc[c] = bfb[c];
    const float* row = feats + (long)n * 256;
    for (int k = 0; k < 256; ++k) {
        float v = row[k];
#pragma unroll
        for (int c = 0; c < NCLS; ++c) acc[c] += v * Wfb[k * NCLS + c];
    }
    float mx = acc[0];
#pragma unroll
    for (int c = 1; c < NCLS; ++c) mx = fmaxf(mx, acc[c]);
    float s = 0.0f;
#pragma unroll
    for (int c = 0; c < NCLS; ++c) s += __expf(acc[c] - mx);
    float lse = mx + __logf(s);
#pragma unroll
    for (int c = 0; c < NCLS; ++c) out[(long)n * NCLS + c] = acc[c] - lse;
}

// ---------------------------------------------------------------------------
extern "C" void kernel_launch(void* const* d_in, const int* in_sizes, int n_in,
                              void* d_out, int out_size, void* d_ws, size_t ws_size,
                              hipStream_t stream) {
    const float* x    = (const float*)d_in[0];
    const int*   ei1  = (const int*)d_in[1];
    const int*   ei2  = (const int*)d_in[2];
    const int*   idx1 = (const int*)d_in[3];
    const int*   idx2 = (const int*)d_in[4];
    const float* W_i1 = (const float*)d_in[5];   const float* b_i1 = (const float*)d_in[6];
    const float* W_i2 = (const float*)d_in[7];   const float* b_i2 = (const float*)d_in[8];
    const float* Wc11 = (const float*)d_in[9];   const float* bc11 = (const float*)d_in[10];
    const float* Wc12 = (const float*)d_in[11];  const float* bc12 = (const float*)d_in[12];
    const float* Wc21 = (const float*)d_in[13];  const float* bc21 = (const float*)d_in[14];
    const float* Wc22 = (const float*)d_in[15];  const float* bc22 = (const float*)d_in[16];
    const float* Wm1a = (const float*)d_in[17];  const float* bm1a = (const float*)d_in[18];
    const float* Wm1b = (const float*)d_in[19];  const float* bm1b = (const float*)d_in[20];
    const float* Wm2a = (const float*)d_in[21];  const float* bm2a = (const float*)d_in[22];
    const float* Wm2b = (const float*)d_in[23];  const float* bm2b = (const float*)d_in[24];
    const float* W_fa = (const float*)d_in[25];  const float* b_fa = (const float*)d_in[26];
    const float* W_fb = (const float*)d_in[27];  const float* b_fb = (const float*)d_in[28];
    float* out = (float*)d_out;

    const int T = T_NODES, N = N_BASE, E = E_EDGES;
    const int* s1 = ei1;      const int* d1 = ei1 + E;
    const int* s2 = ei2;      const int* d2 = ei2 + E;

    // ---- workspace layout: f32 region ----
    float* bufA  = (float*)d_ws;            // T*256 scratch
    float* h     = bufA  + (long)T * 256;   // T*256 main activation
    float* h1    = h     + (long)T * 256;   // T*256 branch / agg 1
    float* h2    = h1    + (long)T * 256;   // T*256 branch / agg 2
    float* dinv1 = h2    + (long)T * 256;   // T
    float* dinv2 = dinv1 + T;               // T
    float* r1    = dinv2 + T;               // N*256
    float* r2    = r1    + (long)N * 256;   // N*256
    float* cnt1  = r2    + (long)N * 256;   // N
    float* cnt2  = cnt1  + N;               // N

    // ---- bf16 region: pre-swizzled weights (16B-aligned by construction) ----
    const long SWK1   = (long)((FIN + 31) / 32) * 8192;   // K=4652
    const long SWK256 = 8L  * 8192;                       // K=256
    const long SWK512 = 16L * 8192;                       // K=512
    __bf16* sw_i1  = (__bf16*)(cnt2 + N);
    __bf16* sw_i2  = sw_i1  + SWK1;
    __bf16* sw_c11 = sw_i2  + SWK256;
    __bf16* sw_c12 = sw_c11 + SWK256;
    __bf16* sw_c21 = sw_c12 + SWK256;
    __bf16* sw_c22 = sw_c21 + SWK256;
    __bf16* sw_m1a = sw_c22 + SWK256;
    __bf16* sw_m1b = sw_m1a + SWK512;
    __bf16* sw_m2a = sw_m1b + SWK256;
    __bf16* sw_m2b = sw_m2a + SWK512;
    __bf16* sw_fa  = sw_m2b + SWK256;

    const int gT   = (T + 127) / 128;
    const int gN5k = (N + 127) / 128;
    const dim3 B256(256), B128(128);

#define SWZ(W_, D_, K_) do { \
        int ns_ = ((K_) + 31) / 32 * 512; \
        hipLaunchKernelGGL(swizzle_w, dim3((ns_ + 255) / 256), B256, 0, stream, \
                           W_, D_, K_, ns_); \
    } while (0)

    SWZ(W_i1, sw_i1, FIN);
    SWZ(W_i2, sw_i2, 256);
    SWZ(Wc11, sw_c11, 256);  SWZ(Wc12, sw_c12, 256);
    SWZ(Wc21, sw_c21, 256);  SWZ(Wc22, sw_c22, 256);
    SWZ(Wm1a, sw_m1a, 512);  SWZ(Wm1b, sw_m1b, 256);
    SWZ(Wm2a, sw_m2a, 512);  SWZ(Wm2b, sw_m2b, 256);
    SWZ(W_fa, sw_fa, 512);

#define GEMM(A1_, A2_, KS_, WSW_, B_, C_, M_, K_, G_, RELU_) \
    hipLaunchKernelGGL(gemm_wmma_bf16, dim3(G_), B256, 0, stream, \
                       A1_, A2_, KS_, WSW_, B_, C_, M_, K_, RELU_)

    // ---- mlp_init: h = relu(x@W_i1+b) @ W_i2 + b ----
    GEMM(x, (const float*)nullptr, 0, sw_i1, b_i1, bufA, T, FIN, gT, 1);
    GEMM(bufA, (const float*)nullptr, 0, sw_i2, b_i2, h, T, 256, gT, 0);

    // ---- symmetric-norm degrees (self-loop included) ----
    hipLaunchKernelGGL(fill_f32, dim3((T + 255) / 256), B256, 0, stream, dinv1, 1.0f, T);
    hipLaunchKernelGGL(fill_f32, dim3((T + 255) / 256), B256, 0, stream, dinv2, 1.0f, T);
    hipLaunchKernelGGL(deg_count, dim3((E + 255) / 256), B256, 0, stream, d1, dinv1, E);
    hipLaunchKernelGGL(deg_count, dim3((E + 255) / 256), B256, 0, stream, d2, dinv2, E);
    hipLaunchKernelGGL(rsqrt_ip, dim3((T + 255) / 256), B256, 0, stream, dinv1, T);
    hipLaunchKernelGGL(rsqrt_ip, dim3((T + 255) / 256), B256, 0, stream, dinv2, T);

#define GCN(SWC_, bc_, sE_, dE_, dinv_, dst_) \
    GEMM(h, (const float*)nullptr, 0, SWC_, (const float*)nullptr, bufA, T, 256, gT, 0); \
    hipLaunchKernelGGL(gcn_self_init, dim3(T), B256, 0, stream, bufA, dinv_, dst_); \
    hipLaunchKernelGGL(gcn_edge_scatter, dim3(E), B256, 0, stream, bufA, sE_, dE_, dinv_, dst_); \
    hipLaunchKernelGGL(bias_relu256, dim3(T), B256, 0, stream, dst_, bc_)

    // ---- round 1 ----
    GCN(sw_c11, bc11, s1, d1, dinv1, h1);
    GCN(sw_c12, bc12, s2, d2, dinv2, h2);
    GEMM(h1, h2, 256, sw_m1a, bm1a, bufA, T, 512, gT, 1);   // concat via split-A
    GEMM(bufA, (const float*)nullptr, 0, sw_m1b, bm1b, h, T, 256, gT, 0);

    // ---- round 2 ----
    GCN(sw_c21, bc21, s1, d1, dinv1, h1);
    GCN(sw_c22, bc22, s2, d2, dinv2, h2);
    GEMM(h1, h2, 256, sw_m2a, bm2a, bufA, T, 512, gT, 1);
    GEMM(bufA, (const float*)nullptr, 0, sw_m2b, bm2b, h, T, 256, gT, 0);

    // ---- scatter-mean to base graph (r1,r2,cnt1,cnt2 contiguous) ----
    hipMemsetAsync(r1, 0, ((size_t)2 * N * 256 + 2 * N) * sizeof(float), stream);
    hipLaunchKernelGGL(seg_count, dim3((T + 255) / 256), B256, 0, stream, idx1, cnt1, T);
    hipLaunchKernelGGL(seg_count, dim3((T + 255) / 256), B256, 0, stream, idx2, cnt2, T);
    hipLaunchKernelGGL(seg_sum256, dim3(T), B256, 0, stream, h, idx1, r1);
    hipLaunchKernelGGL(seg_sum256, dim3(T), B256, 0, stream, h, idx2, r2);
    hipLaunchKernelGGL(seg_div256, dim3(N), B256, 0, stream, r1, cnt1);
    hipLaunchKernelGGL(seg_div256, dim3(N), B256, 0, stream, r2, cnt2);

    // ---- final mlp: relu([r1 r2]@W_fa+b) @ W_fb + b, log_softmax ----
    GEMM(r1, r2, 256, sw_fa, b_fa, bufA, N, 512, gN5k, 1);
    hipLaunchKernelGGL(final_head, dim3((N + 127) / 128), B128, 0, stream,
                       bufA, W_fb, b_fb, out, N);

#undef GCN
#undef GEMM
#undef SWZ
}
